// Net_RNN_11390253269736
// MI455X (gfx1250) — matrix operand order repaired and checked by reading, not compiled, not run on hardware
//
#include <hip/hip_runtime.h>
#include <hip/hip_fp16.h>

typedef __attribute__((ext_vector_type(16))) _Float16 v16h;
typedef __attribute__((ext_vector_type(8)))  _Float16 v8h;
typedef __attribute__((ext_vector_type(8)))  float    v8f;

// ---------------------------------------------------------------------------
// Graph-norm precompute
// ---------------------------------------------------------------------------
__global__ void k_deg_init(float* deg, int n) {
    int i = blockIdx.x * blockDim.x + threadIdx.x;
    if (i < n) deg[i] = 1.0f;                       // self-loop
}

__global__ void k_deg_edges(const int* __restrict__ ei, float* deg, int E) {
    int e = blockIdx.x * blockDim.x + threadIdx.x;
    if (e < E) unsafeAtomicAdd(&deg[ei[E + e]], 1.0f);   // dst side
}

__global__ void k_dis(float* deg_dis, float* nself, int n) {
    int i = blockIdx.x * blockDim.x + threadIdx.x;
    if (i >= n) return;
    float d = deg_dis[i];
    float r = (d > 0.f) ? rsqrtf(d) : 0.f;
    deg_dis[i] = r;             // in-place: deg -> dis
    nself[i]   = r * r;         // self-loop norm
}

__global__ void k_enorm(const int* __restrict__ ei, const float* __restrict__ dis,
                        float* enorm, int E) {
    int e = blockIdx.x * blockDim.x + threadIdx.x;
    if (e < E) enorm[e] = dis[ei[e]] * dis[ei[E + e]];
}

// ---------------------------------------------------------------------------
// Weight conversion: [32, cout] f32 row-major -> [32, 32] f16, zero padded
// ---------------------------------------------------------------------------
__global__ void k_convw(const float* __restrict__ W, _Float16* Wh, int cout) {
    int i = blockIdx.x * blockDim.x + threadIdx.x;
    if (i >= 1024) return;
    int k = i >> 5, n = i & 31;
    Wh[i] = (n < cout) ? (_Float16)W[k * cout + n] : (_Float16)0.f;
}

// ---------------------------------------------------------------------------
// Initial state: h0 = relu(y0 @ fc1 + b1); h = relu(h0 @ fc2 + b2) [N,26]
// also out[0:N] = y[0]
// ---------------------------------------------------------------------------
__global__ void k_init_h(const float* __restrict__ y,
                         const float* __restrict__ fc1W, const float* __restrict__ fc1b,
                         const float* __restrict__ fc2W, const float* __restrict__ fc2b,
                         float* __restrict__ h3, float* __restrict__ out, int n) {
    int i = blockIdx.x * blockDim.x + threadIdx.x;
    if (i >= n) return;
    float yv = y[i];            // y row 0
    out[i] = yv;
    float h0[32];
    #pragma unroll
    for (int j = 0; j < 32; ++j) {
        float v = yv * fc1W[j] + fc1b[j];
        h0[j] = v > 0.f ? v : 0.f;
    }
    for (int c = 0; c < 32; ++c) {
        float acc = 0.f;
        if (c < 26) {
            acc = fc2b[c];
            #pragma unroll
            for (int j = 0; j < 32; ++j) acc += h0[j] * fc2W[j * 26 + c];
            acc = acc > 0.f ? acc : 0.f;
        }
        h3[(size_t)i * 32 + c] = acc;
    }
}

// ---------------------------------------------------------------------------
// hin = concat(h[:, :26], x3, t3) -> f16 [N, 32]
// ---------------------------------------------------------------------------
__global__ void k_build_hin(const float* __restrict__ h3, const float* __restrict__ x,
                            const float* __restrict__ t, int tt,
                            _Float16* __restrict__ hA, int n) {
    int tid = blockIdx.x * blockDim.x + threadIdx.x;
    if (tid >= n * 32) return;
    int i = tid >> 5, j = tid & 31;
    float v;
    if (j < 26)      v = h3[tid];
    else if (j < 29) v = x[i];
    else             v = t[tt];
    hA[tid] = (_Float16)v;
}

// ---------------------------------------------------------------------------
// WMMA GEMM: z = A[N,32] @ W[32,32], agg = z * nself (self-loop init)
// One wave per 16-row tile; two v_wmma_f32_16x16x32_f16 per wave.
// ---------------------------------------------------------------------------
__global__ __launch_bounds__(256)
void k_gemm16(const _Float16* __restrict__ A, const _Float16* __restrict__ Wh,
              const float* __restrict__ nself,
              float* __restrict__ z, float* __restrict__ agg, int ntiles) {
    __shared__ _Float16 sW[1024];
    for (int i = threadIdx.x; i < 1024; i += 256) sW[i] = Wh[i];
    __syncthreads();

    const int wave = threadIdx.x >> 5;
    const int lane = threadIdx.x & 31;
    const int tile = blockIdx.x * 8 + wave;
    if (tile >= ntiles) return;              // wave-uniform: EXEC stays all-ones

    const int hi = lane >> 4;                // half-wave select
    const int ln = lane & 15;
    const int rowbase = tile << 4;

    // A fragment (16-bit 16x32 layout): lane<16 holds K 0..7,16..23; lane>=16: 8..15,24..31
    const _Float16* arow = A + (size_t)(rowbase + ln) * 32;
    v8h alo = *(const v8h*)(arow + hi * 8);
    v8h ahi = *(const v8h*)(arow + 16 + hi * 8);
    v16h a;
    #pragma unroll
    for (int i = 0; i < 8; ++i) { a[i] = alo[i]; a[8 + i] = ahi[i]; }

    const int r0 = rowbase + hi * 8;         // D rows this lane owns
    float ns[8];
    #pragma unroll
    for (int v = 0; v < 8; ++v) ns[v] = nself[r0 + v];

    #pragma unroll
    for (int g = 0; g < 2; ++g) {
        const int col = g * 16 + ln;
        // B fragment (32x16): lane holds column `ln`, K = hi*8..hi*8+7, 16+hi*8..16+hi*8+7
        v16h b;
        #pragma unroll
        for (int i = 0; i < 16; ++i) b[i] = sW[(hi * 8 + i + (i & 8)) * 32 + col];

        v8f c = {0.f, 0.f, 0.f, 0.f, 0.f, 0.f, 0.f, 0.f};
        v8f d = __builtin_amdgcn_wmma_f32_16x16x32_f16(
            /*neg_a=*/false, a, /*neg_b=*/false, b,
            /*c_mod=*/(short)0, c, /*reuse_a=*/false, /*reuse_b=*/false);

        // D layout: lane<16 -> rows r0..r0+7 of col, lane>=16 -> rows 8..15
        #pragma unroll
        for (int v = 0; v < 8; ++v) {
            const size_t o = (size_t)(r0 + v) * 32 + col;
            z[o]   = d[v];
            agg[o] = d[v] * ns[v];
        }
    }
}

// ---------------------------------------------------------------------------
// Edge aggregation: agg[dst] += z[src] * enorm[e]; 8 lanes/edge x float4
// ---------------------------------------------------------------------------
__global__ void k_edge_agg(const int* __restrict__ ei, const float* __restrict__ enorm,
                           const float* __restrict__ z, float* __restrict__ agg, int E) {
    long long tid = (long long)blockIdx.x * blockDim.x + threadIdx.x;
    int e = (int)(tid >> 3);
    if (e >= E) return;
    int c = ((int)tid & 7) * 4;
    int s = ei[e], d = ei[E + e];
    float w = enorm[e];
    const float4 v = *(const float4*)(z + (size_t)s * 32 + c);
    float* ap = agg + (size_t)d * 32 + c;
    unsafeAtomicAdd(ap + 0, v.x * w);
    unsafeAtomicAdd(ap + 1, v.y * w);
    unsafeAtomicAdd(ap + 2, v.z * w);
    unsafeAtomicAdd(ap + 3, v.w * w);
}

// ---------------------------------------------------------------------------
// h = relu(agg + bias); write f16 (next GEMM input) and optional f32 state
// ---------------------------------------------------------------------------
__global__ void k_bias_relu(const float* __restrict__ agg, const float* __restrict__ b,
                            int cout, _Float16* __restrict__ hA,
                            float* __restrict__ h3opt, int n) {
    int tid = blockIdx.x * blockDim.x + threadIdx.x;
    if (tid >= n * 32) return;
    int j = tid & 31;
    float v = 0.f;
    if (j < cout) {
        v = agg[tid] + b[j];
        v = v > 0.f ? v : 0.f;
    }
    hA[tid] = (_Float16)v;
    if (h3opt) h3opt[tid] = v;
}

// ---------------------------------------------------------------------------
// Readout: out = relu(h3[:, :26] @ fc3 + b3) @ fc4 + b4
// ---------------------------------------------------------------------------
__global__ void k_readout(const float* __restrict__ h3,
                          const float* __restrict__ fc3W, const float* __restrict__ fc3b,
                          const float* __restrict__ fc4W, const float* __restrict__ fc4b,
                          float* __restrict__ out, int n) {
    int i = blockIdx.x * blockDim.x + threadIdx.x;
    if (i >= n) return;
    const float* hr = h3 + (size_t)i * 32;
    float acc = fc4b[0];
    for (int j = 0; j < 32; ++j) {
        float g = fc3b[j];
        #pragma unroll
        for (int k = 0; k < 26; ++k) g += hr[k] * fc3W[k * 32 + j];
        g = g > 0.f ? g : 0.f;
        acc += g * fc4W[j];
    }
    out[i] = acc;
}

// ---------------------------------------------------------------------------
extern "C" void kernel_launch(void* const* d_in, const int* in_sizes, int n_in,
                              void* d_out, int out_size, void* d_ws, size_t ws_size,
                              hipStream_t stream) {
    const float* x     = (const float*)d_in[0];
    const float* t     = (const float*)d_in[1];
    const float* y     = (const float*)d_in[2];
    const int*   ei    = (const int*)  d_in[3];
    const float* fc1W  = (const float*)d_in[4];
    const float* fc1b  = (const float*)d_in[5];
    const float* fc2W  = (const float*)d_in[6];
    const float* fc2b  = (const float*)d_in[7];
    const float* c1W   = (const float*)d_in[8];
    const float* c1b   = (const float*)d_in[9];
    const float* c2W   = (const float*)d_in[10];
    const float* c2b   = (const float*)d_in[11];
    const float* c3W   = (const float*)d_in[12];
    const float* c3b   = (const float*)d_in[13];
    const float* fc3W  = (const float*)d_in[14];
    const float* fc3b  = (const float*)d_in[15];
    const float* fc4W  = (const float*)d_in[16];
    const float* fc4b  = (const float*)d_in[17];

    const int N = in_sizes[0];          // 100000
    const int T = in_sizes[1];          // 50
    const int E = in_sizes[3] / 2;      // 1600000
    float* out = (float*)d_out;

    const int ntiles = (N + 15) / 16;
    const int Np     = ntiles * 16;     // padded row count

    // workspace carve-out
    char* ws = (char*)d_ws;
    auto carve = [&](size_t bytes) {
        char* p = ws;
        ws += (bytes + 255) & ~(size_t)255;
        return p;
    };
    float*    dis   = (float*)   carve((size_t)Np * 4);        // deg -> dis
    float*    nself = (float*)   carve((size_t)Np * 4);
    float*    enorm = (float*)   carve((size_t)E * 4);
    _Float16* hA    = (_Float16*)carve((size_t)Np * 32 * 2);
    float*    z     = (float*)   carve((size_t)Np * 32 * 4);
    float*    agg   = (float*)   carve((size_t)Np * 32 * 4);
    float*    h3    = (float*)   carve((size_t)Np * 32 * 4);
    _Float16* W1h   = (_Float16*)carve(1024 * 2);
    _Float16* W2h   = (_Float16*)carve(1024 * 2);
    _Float16* W3h   = (_Float16*)carve(1024 * 2);

    const int B  = 256;
    const int gN  = (N + B - 1) / B;
    const int gNp = (Np + B - 1) / B;
    const int gE  = (E + B - 1) / B;
    const int gEl = (int)(((long long)E * 8 + B - 1) / B);   // edge-agg grid
    const int gF  = (N * 32 + B - 1) / B;                    // feature-elt grid
    const int gG  = (ntiles + 7) / 8;                        // gemm grid (8 waves/blk)

    // ---- precompute ----
    k_deg_init <<<gNp, B, 0, stream>>>(dis, Np);
    k_deg_edges<<<gE,  B, 0, stream>>>(ei, dis, E);
    k_dis      <<<gNp, B, 0, stream>>>(dis, nself, Np);
    k_enorm    <<<gE,  B, 0, stream>>>(ei, dis, enorm, E);
    k_convw    <<<4,   B, 0, stream>>>(c1W, W1h, 32);
    k_convw    <<<4,   B, 0, stream>>>(c2W, W2h, 32);
    k_convw    <<<4,   B, 0, stream>>>(c3W, W3h, 26);
    k_init_h   <<<gN,  B, 0, stream>>>(y, fc1W, fc1b, fc2W, fc2b, h3, out, N);

    // ---- time recurrence ----
    for (int tt = 1; tt < T; ++tt) {
        k_build_hin<<<gF, B, 0, stream>>>(h3, x, t, tt, hA, N);

        // layer 1
        k_gemm16   <<<gG,  B, 0, stream>>>(hA, W1h, nself, z, agg, ntiles);
        k_edge_agg <<<gEl, B, 0, stream>>>(ei, enorm, z, agg, E);
        k_bias_relu<<<gF,  B, 0, stream>>>(agg, c1b, 32, hA, nullptr, N);

        // layer 2
        k_gemm16   <<<gG,  B, 0, stream>>>(hA, W2h, nself, z, agg, ntiles);
        k_edge_agg <<<gEl, B, 0, stream>>>(ei, enorm, z, agg, E);
        k_bias_relu<<<gF,  B, 0, stream>>>(agg, c2b, 32, hA, nullptr, N);

        // layer 3 (26 real output channels, padded to 32)
        k_gemm16   <<<gG,  B, 0, stream>>>(hA, W3h, nself, z, agg, ntiles);
        k_edge_agg <<<gEl, B, 0, stream>>>(ei, enorm, z, agg, E);
        k_bias_relu<<<gF,  B, 0, stream>>>(agg, c3b, 26, hA, h3, N);

        k_readout  <<<gN,  B, 0, stream>>>(h3, fc3W, fc3b, fc4W, fc4b,
                                           out + (size_t)tt * N, N);
    }
}
